// stGCL_29850022707202
// MI455X (gfx1250) — matrix operand order repaired and checked by reading, not compiled, not run on hardware
//
#include <hip/hip_runtime.h>
#include <hip/hip_bf16.h>
#include <stdint.h>

// ---------------------------------------------------------------------------
// Types for CDNA5 WMMA (wave32, v_wmma_f32_16x16x32_bf16)
// ---------------------------------------------------------------------------
typedef __attribute__((ext_vector_type(16))) __bf16        v16bf;
typedef __attribute__((ext_vector_type(8)))  float         v8f;
typedef __attribute__((ext_vector_type(4)))  unsigned int  u32x4;

// gfx1250 async global->LDS path (ASYNCcnt), with graceful fallback.
// Builtin signature (from hipcc diagnostics): param0 = int4(vector_size 16)
// pointer in AS(1) (global), param1 = int4 pointer in AS(3) (LDS), then two
// integer immediates (offset, cpol).
#if __has_builtin(__builtin_amdgcn_global_load_async_to_lds_b128) && \
    __has_builtin(__builtin_amdgcn_s_wait_asynccnt)
#define USE_ASYNC_LDS 1
typedef int v4i_gcc __attribute__((vector_size(16)));
typedef __attribute__((address_space(1))) v4i_gcc gv4i;
typedef __attribute__((address_space(3))) v4i_gcc lv4i;
#else
#define USE_ASYNC_LDS 0
#endif

#define NEG_SLOPE 0.2f
#define BM 128
#define BN 64
#define BK 32
#define GEMM_THREADS 256

// ---------------------------------------------------------------------------
// Helpers
// ---------------------------------------------------------------------------
__device__ __forceinline__ unsigned short f32_to_bf16_rne(float f) {
  unsigned u = __float_as_uint(f);
  u += 0x7fffu + ((u >> 16) & 1u);   // round-to-nearest-even
  return (unsigned short)(u >> 16);
}
__device__ __forceinline__ unsigned f2ord(float f) {
  unsigned u = __float_as_uint(f);
  return (u & 0x80000000u) ? ~u : (u | 0x80000000u);
}
__device__ __forceinline__ float ord2f(unsigned o) {
  unsigned u = (o & 0x80000000u) ? (o & 0x7fffffffu) : ~o;
  return __uint_as_float(u);
}

// ---------------------------------------------------------------------------
// Elementwise kernels
// ---------------------------------------------------------------------------
__global__ void zero_f32_kernel(float* __restrict__ p, long n) {
  long i = (long)blockIdx.x * blockDim.x + threadIdx.x;
  long stride = (long)gridDim.x * blockDim.x;
  for (; i < n; i += stride) p[i] = 0.0f;
}

__global__ void cvt_bf16_kernel(const float* __restrict__ in,
                                unsigned short* __restrict__ out, long n) {
  long i = (long)blockIdx.x * blockDim.x + threadIdx.x;
  long stride = (long)gridDim.x * blockDim.x;
  for (; i < n; i += stride) out[i] = f32_to_bf16_rne(in[i]);
}

// Transposed bf16 convert: in is [Kd x Nd] row-major f32, out is [Nd x Kd] bf16.
__global__ void cvt_bf16_T_kernel(const float* __restrict__ in,
                                  unsigned short* __restrict__ out,
                                  int Kd, int Nd) {
  long total = (long)Kd * Nd;
  long i = (long)blockIdx.x * blockDim.x + threadIdx.x;
  long stride = (long)gridDim.x * blockDim.x;
  for (; i < total; i += stride) {
    long k = i / Nd;
    int  n = (int)(i - k * Nd);
    out[(long)n * Kd + k] = f32_to_bf16_rne(in[i]);
  }
}

// ELU fused with bf16 store: out[r*ld_out + col_off + c] = bf16(elu(in[r*D+c]))
__global__ void elu_bf16_kernel(const float* __restrict__ in,
                                unsigned short* __restrict__ out,
                                long N, int D, int ld_out, int col_off) {
  long total = N * (long)D;
  long i = (long)blockIdx.x * blockDim.x + threadIdx.x;
  long stride = (long)gridDim.x * blockDim.x;
  for (; i < total; i += stride) {
    long r = i / D;
    int  c = (int)(i - r * D);
    float v = in[i];
    v = (v > 0.0f) ? v : (__expf(v) - 1.0f);
    out[r * (long)ld_out + col_off + c] = f32_to_bf16_rne(v);
  }
}

// ---------------------------------------------------------------------------
// Tiled bf16 WMMA GEMM:  C[M x N] = A[M x K] * B^T-layout
//   A: bf16(u16), row stride lda; optional rowmap gathers A's rows.
//   B: bf16(u16) stored k-contiguous: element (k,n) at B[n*ldb + k].
//   C: f32, row-major ldc = N.
// Block tile 128x64, BK=32, 256 threads = 8 waves (2x2 WMMA each).
// Full K-tiles stream through a double-buffered LDS pipeline using
// GLOBAL_LOAD_ASYNC_TO_LDS_B128 (ASYNCcnt) when available; only the K-tail
// uses a guarded scalar path with zero fill.
// ---------------------------------------------------------------------------
__global__ __launch_bounds__(GEMM_THREADS)
void gemm_bf16_wmma_kernel(const unsigned short* __restrict__ A, int lda,
                           const int* __restrict__ rowmap,
                           const unsigned short* __restrict__ B, int ldb,
                           float* __restrict__ C, int M, int N, int K) {
  __shared__ unsigned short As[2][BM * BK];   // [m][k]
  __shared__ unsigned short Bst[2][BN * BK];  // [n][k]

  const int t    = threadIdx.x;
  const int m0   = blockIdx.y * BM;
  const int n0   = blockIdx.x * BN;
  const int lane = t & 31;
  const int wid  = t >> 5;
  const int wm   = (wid >> 1) * 32;
  const int wn   = (wid & 1)  * 32;
  const int half = lane >> 4;
  const int l15  = lane & 15;

  const v8f vzero = {0.f, 0.f, 0.f, 0.f, 0.f, 0.f, 0.f, 0.f};
  v8f acc[2][2];
#pragma unroll
  for (int i = 0; i < 2; ++i)
#pragma unroll
    for (int j = 0; j < 2; ++j) acc[i][j] = vzero;

  // A staging: thread t -> row t>>1 (0..127), k-segment (t&1)*16 (32B = 2xb128)
  const int  a_row = t >> 1;
  const int  a_ks  = (t & 1) * 16;
  const bool a_ok  = (m0 + a_row) < M;
  const long a_src = a_ok ? (rowmap ? (long)rowmap[m0 + a_row] : (long)(m0 + a_row)) : 0;

  // B staging: thread t -> col t>>2 (0..63), k-segment (t&3)*8 (16B = 1xb128)
  const int  b_col = t >> 2;
  const int  b_ks  = (t & 3) * 8;
  const int  b_n   = n0 + b_col;
  const bool b_ok  = b_n < N;
  const long b_src = b_ok ? (long)b_n : 0;

  // Stage a full (no bounds needed) K-tile into buffer `buf`.
  auto stage_full = [&](int buf, int k0) {
    const unsigned short* ga = A + a_src * lda + k0 + a_ks;
    unsigned short*       la = &As[buf][a_row * BK + a_ks];
    const unsigned short* gb = B + b_src * ldb + k0 + b_ks;
    unsigned short*       lb = &Bst[buf][b_col * BK + b_ks];
#if USE_ASYNC_LDS
    __builtin_amdgcn_global_load_async_to_lds_b128((gv4i*)ga,       (lv4i*)la,       0, 0);
    __builtin_amdgcn_global_load_async_to_lds_b128((gv4i*)(ga + 8), (lv4i*)(la + 8), 0, 0);
    __builtin_amdgcn_global_load_async_to_lds_b128((gv4i*)gb,       (lv4i*)lb,       0, 0);
#else
    __builtin_prefetch(ga + BK, 0, 0);   // global_prefetch_b8 for next tile
    u32x4 va0 = *(const u32x4*)(const void*)ga;
    u32x4 va1 = *(const u32x4*)(const void*)(ga + 8);
    u32x4 vb  = *(const u32x4*)(const void*)gb;
    *(u32x4*)(void*)la       = va0;
    *(u32x4*)(void*)(la + 8) = va1;
    *(u32x4*)(void*)lb       = vb;
#endif
  };

  // Guarded, zero-filling staging for the K-tail.
  auto stage_tail = [&](int buf, int k0) {
#pragma unroll
    for (int i = 0; i < 16; ++i) {
      int k = k0 + a_ks + i;
      As[buf][a_row * BK + a_ks + i] =
          (a_ok && k < K) ? A[a_src * lda + k] : (unsigned short)0;
    }
#pragma unroll
    for (int i = 0; i < 8; ++i) {
      int k = k0 + b_ks + i;
      Bst[buf][b_col * BK + b_ks + i] =
          (b_ok && k < K) ? B[b_src * ldb + k] : (unsigned short)0;
    }
  };

  // 4x v_wmma_f32_16x16x32_bf16 on the staged tile.
  auto compute_tile = [&](int buf) {
    union FragU { u32x4 q[2]; v16bf v; };
    FragU afrag[2], bfrag[2];
    const unsigned short* As_  = &As[buf][0];
    const unsigned short* Bst_ = &Bst[buf][0];
#pragma unroll
    for (int i = 0; i < 2; ++i) {
      const u32x4* pa =
          (const u32x4*)(const void*)(As_ + (wm + 16 * i + l15) * BK + 8 * half);
      afrag[i].q[0] = pa[0];  // k = 8h .. 8h+7
      afrag[i].q[1] = pa[2];  // k = 16+8h .. 16+8h+7
      const u32x4* pb =
          (const u32x4*)(const void*)(Bst_ + (wn + 16 * i + l15) * BK + 8 * half);
      bfrag[i].q[0] = pb[0];  // k = 8h .. 8h+7
      bfrag[i].q[1] = pb[2];  // k = 16+8h .. 16+8h+7
    }
#pragma unroll
    for (int i = 0; i < 2; ++i)
#pragma unroll
      for (int j = 0; j < 2; ++j)
        acc[i][j] = __builtin_amdgcn_wmma_f32_16x16x32_bf16(
            false, afrag[i].v, false, bfrag[j].v, (short)0, acc[i][j],
            false, false);
  };

  const int nfull = K / BK;
  const int ktail = K - nfull * BK;

  if (nfull > 0) {
    stage_full(0, 0);
    for (int it = 0; it < nfull; ++it) {
      const int buf = it & 1;
      if (it + 1 < nfull) {
        stage_full(buf ^ 1, (it + 1) * BK);   // overlap next tile with compute
#if USE_ASYNC_LDS
        __builtin_amdgcn_s_wait_asynccnt(3);  // older stage (3 b128s) landed
#endif
      } else {
#if USE_ASYNC_LDS
        __builtin_amdgcn_s_wait_asynccnt(0);
#endif
      }
      __syncthreads();
      compute_tile(buf);
      __syncthreads();
    }
  }
  if (ktail) {
    const int tb = nfull & 1;
    stage_tail(tb, nfull * BK);
    __syncthreads();
    compute_tile(tb);
  }

  // D layout: lane (half,l15): VGPR r -> row = r + 8*half, col = l15
#pragma unroll
  for (int i = 0; i < 2; ++i)
#pragma unroll
    for (int j = 0; j < 2; ++j)
#pragma unroll
      for (int r = 0; r < 8; ++r) {
        int row = m0 + wm + 16 * i + r + 8 * half;
        int col = n0 + wn + 16 * j + l15;
        if (row < M && col < N) C[(long)row * N + col] = acc[i][j][r];
      }
}

// ---------------------------------------------------------------------------
// GAT attention kernels
// ---------------------------------------------------------------------------
__global__ void attn_scores_kernel(const float* __restrict__ H,
                                   const float* __restrict__ a_s,
                                   const float* __restrict__ a_d,
                                   float* __restrict__ es, float* __restrict__ ed,
                                   int N, int D) {
  int gw   = (int)(((long)blockIdx.x * blockDim.x + threadIdx.x) >> 5);
  int lane = threadIdx.x & 31;
  if (gw >= N) return;
  const float* h = H + (long)gw * D;
  float s = 0.0f, d = 0.0f;
  for (int i = lane; i < D; i += 32) {
    float v = h[i];
    s += v * a_s[i];
    d += v * a_d[i];
  }
#pragma unroll
  for (int o = 16; o > 0; o >>= 1) {
    s += __shfl_down(s, o, 32);
    d += __shfl_down(d, o, 32);
  }
  if (lane == 0) { es[gw] = s; ed[gw] = d; }
}

__global__ void edge_logits_max_kernel(const float* __restrict__ es,
                                       const float* __restrict__ ed,
                                       const int* __restrict__ src,
                                       const int* __restrict__ dst,
                                       float* __restrict__ e,
                                       unsigned* __restrict__ mkey, int E) {
  int i = blockIdx.x * blockDim.x + threadIdx.x;
  if (i >= E) return;
  float v = es[src[i]] + ed[dst[i]];
  v = (v >= 0.0f) ? v : NEG_SLOPE * v;
  e[i] = v;
  atomicMax(&mkey[dst[i]], f2ord(v));
}

__global__ void edge_exp_sum_kernel(float* __restrict__ e,
                                    const unsigned* __restrict__ mkey,
                                    const int* __restrict__ dst,
                                    float* __restrict__ ssum, int E) {
  int i = blockIdx.x * blockDim.x + threadIdx.x;
  if (i >= E) return;
  float m  = ord2f(mkey[dst[i]]);
  float ex = __expf(e[i] - m);
  e[i] = ex;
  atomicAdd(&ssum[dst[i]], ex);
}

__global__ void edge_aggregate_kernel(const float* __restrict__ H,
                                      const float* __restrict__ ex,
                                      const float* __restrict__ ssum,
                                      const int* __restrict__ src,
                                      const int* __restrict__ dst,
                                      float* __restrict__ out, int E, int D) {
  int gw   = (int)(((long)blockIdx.x * blockDim.x + threadIdx.x) >> 5);
  int lane = threadIdx.x & 31;
  if (gw >= E) return;
  int s = src[gw], d = dst[gw];
  float coef = ex[gw] / (ssum[d] + 1e-16f);
  const float* hs = H + (long)s * D;
  float* od = out + (long)d * D;
  for (int i = lane; i < D; i += 32) atomicAdd(&od[i], hs[i] * coef);
}

__global__ void mean_sigmoid_kernel(const float* __restrict__ h2,
                                    float* __restrict__ out, int N, int D) {
  int c = threadIdx.x;
  if (c >= D) return;
  float s = 0.0f;
  for (int r = 0; r < N; ++r) s += h2[(long)r * D + c];
  s /= (float)N;
  out[c] = 1.0f / (1.0f + __expf(-s));
}

// ---------------------------------------------------------------------------
// Host orchestration
// ---------------------------------------------------------------------------
extern "C" void kernel_launch(void* const* d_in, const int* in_sizes, int n_in,
                              void* d_out, int out_size, void* d_ws, size_t ws_size,
                              hipStream_t stream) {
  (void)n_in; (void)out_size; (void)ws_size;

  const float* features    = (const float*)d_in[0];
  const float* im_features = (const float*)d_in[1];
  const float* W1x         = (const float*)d_in[2];
  const float* a1x_src     = (const float*)d_in[3];
  const float* a1x_dst     = (const float*)d_in[4];
  const float* W1r         = (const float*)d_in[5];
  const float* a1r_src     = (const float*)d_in[6];
  const float* a1r_dst     = (const float*)d_in[7];
  const float* W2          = (const float*)d_in[8];
  const float* a3_src      = (const float*)d_in[9];
  const float* a3_dst      = (const float*)d_in[10];
  const int*   edge_index  = (const int*)d_in[11];
  const int*   perm        = (const int*)d_in[12];

  const int NH   = in_sizes[3];            // 256
  const int IN_X = in_sizes[2] / NH;       // 3000
  const int IN_R = in_sizes[5] / NH;       // 2048
  const int OUT  = in_sizes[8] / (2 * NH); // 64
  const int E    = in_sizes[11] / 2;       // 480000
  const int Nn   = in_sizes[0] / IN_X;     // 30000
  const int D2   = 2 * NH;                 // 512
  const int* src = edge_index;
  const int* dst = edge_index + E;

  // ---- workspace carve ----
  char* w = (char*)d_ws;
  size_t off = 0;
  auto carve = [&](size_t bytes) -> char* {
    char* p = w + off;
    off = (off + bytes + 255) & ~(size_t)255;
    return p;
  };
  unsigned short* Xbf   = (unsigned short*)carve((size_t)Nn * IN_X * 2);
  unsigned short* Rbf   = (unsigned short*)carve((size_t)Nn * IN_R * 2);
  unsigned short* W1xbf = (unsigned short*)carve((size_t)IN_X * NH * 2); // [IN_X][NH]
  unsigned short* W1rbf = (unsigned short*)carve((size_t)IN_R * NH * 2); // [IN_R][NH]
  unsigned short* W2bf  = (unsigned short*)carve((size_t)D2 * OUT * 2);  // [D2][OUT]
  unsigned short* W1xT  = (unsigned short*)carve((size_t)NH * IN_X * 2); // [NH][IN_X]
  unsigned short* W1rT  = (unsigned short*)carve((size_t)NH * IN_R * 2); // [NH][IN_R]
  unsigned short* W2T   = (unsigned short*)carve((size_t)OUT * D2 * 2);  // [OUT][D2]
  float* Hpre  = (float*)carve((size_t)Nn * D2 * 4);
  float* Hagg  = (float*)carve((size_t)Nn * D2 * 4);
  unsigned short* Cat  = (unsigned short*)carve((size_t)Nn * D2 * 2);
  unsigned short* H2bf = (unsigned short*)carve((size_t)Nn * OUT * 2);
  float*    es   = (float*)carve((size_t)Nn * 4);
  float*    ed   = (float*)carve((size_t)Nn * 4);
  float*    elog = (float*)carve((size_t)E * 4);
  unsigned* mkey = (unsigned*)carve((size_t)Nn * 4);
  float*    ssum = (float*)carve((size_t)Nn * 4);

  // ---- output slots ----
  float* out = (float*)d_out;
  size_t oh2   = 0;
  size_t oh4x  = oh2   + (size_t)Nn * OUT;
  size_t oh4r  = oh4x  + (size_t)Nn * IN_X;
  size_t orh2  = oh4r  + (size_t)Nn * IN_R;
  size_t orh4x = orh2  + (size_t)Nn * OUT;
  size_t orh4r = orh4x + (size_t)Nn * IN_X;
  size_t osum  = orh4r + (size_t)Nn * IN_R;

  // ---- launch helpers ----
  auto zero = [&](void* p, long n_f32) {
    long b = (n_f32 + 255) / 256;
    int blocks = (int)(b < 4096 ? b : 4096);
    if (blocks < 1) blocks = 1;
    zero_f32_kernel<<<blocks, 256, 0, stream>>>((float*)p, n_f32);
  };
  auto cvt = [&](const float* in, unsigned short* o, long n) {
    long b = (n + 255) / 256;
    int blocks = (int)(b < 8192 ? b : 8192);
    cvt_bf16_kernel<<<blocks, 256, 0, stream>>>(in, o, n);
  };
  auto cvtT = [&](const float* in, unsigned short* o, int Kd, int Nd) {
    long n = (long)Kd * Nd;
    long b = (n + 255) / 256;
    int blocks = (int)(b < 8192 ? b : 8192);
    cvt_bf16_T_kernel<<<blocks, 256, 0, stream>>>(in, o, Kd, Nd);
  };
  auto gemm = [&](const unsigned short* A, int lda, const int* rowmap,
                  const unsigned short* B, int ldb,
                  float* C, int M, int Ncols, int K) {
    dim3 g((Ncols + BN - 1) / BN, (M + BM - 1) / BM);
    gemm_bf16_wmma_kernel<<<g, GEMM_THREADS, 0, stream>>>(
        A, lda, rowmap, B, ldb, C, M, Ncols, K);
  };
  auto gat = [&](const float* H, const float* a_s, const float* a_d,
                 float* agg, int D) {
    zero(mkey, Nn);
    zero(ssum, Nn);
    zero(agg, (long)Nn * D);
    attn_scores_kernel<<<(Nn + 7) / 8, 256, 0, stream>>>(H, a_s, a_d, es, ed, Nn, D);
    edge_logits_max_kernel<<<(E + 255) / 256, 256, 0, stream>>>(es, ed, src, dst,
                                                                elog, mkey, E);
    edge_exp_sum_kernel<<<(E + 255) / 256, 256, 0, stream>>>(elog, mkey, dst, ssum, E);
    edge_aggregate_kernel<<<(E + 7) / 8, 256, 0, stream>>>(H, elog, ssum, src, dst,
                                                           agg, E, D);
  };
  auto elu2bf = [&](const float* in, unsigned short* o, int D, int ld, int coff) {
    long n = (long)Nn * D;
    long b = (n + 255) / 256;
    int blocks = (int)(b < 8192 ? b : 8192);
    elu_bf16_kernel<<<blocks, 256, 0, stream>>>(in, o, Nn, D, ld, coff);
  };

  // ---- one-time bf16 conversions (plus k-contiguous transposed weights) ----
  cvt(features,    Xbf,   (long)Nn * IN_X);
  cvt(im_features, Rbf,   (long)Nn * IN_R);
  cvt(W1x,         W1xbf, (long)IN_X * NH);
  cvt(W1r,         W1rbf, (long)IN_R * NH);
  cvt(W2,          W2bf,  (long)D2 * OUT);
  cvtT(W1x, W1xT, IN_X, NH);   // [NH][IN_X]
  cvtT(W1r, W1rT, IN_R, NH);   // [NH][IN_R]
  cvtT(W2,  W2T,  D2,  OUT);   // [OUT][D2]

  // ---- encode: h2 = (elu(GATx) || elu(GATr)) @ W2 ----
  auto encode = [&](const int* rowmap, float* h2_out) {
    gemm(Xbf, IN_X, rowmap, W1xT, IN_X, Hpre, Nn, NH, IN_X);
    gat(Hpre, a1x_src, a1x_dst, Hagg, NH);
    elu2bf(Hagg, Cat, NH, D2, 0);
    gemm(Rbf, IN_R, rowmap, W1rT, IN_R, Hpre, Nn, NH, IN_R);
    gat(Hpre, a1r_src, a1r_dst, Hagg, NH);
    elu2bf(Hagg, Cat, NH, D2, NH);
    gemm(Cat, D2, nullptr, W2T, D2, h2_out, Nn, OUT, D2);
  };

  // ---- decode: h3 = elu(GAT(h2 @ W2.T)); h4x = k1@W1x.T; h4r = k2@W1r.T ----
  auto decode = [&](const float* h2_f32, float* h4x, float* h4r) {
    cvt(h2_f32, H2bf, (long)Nn * OUT);
    gemm(H2bf, OUT, nullptr, W2bf, OUT, Hpre, Nn, D2, OUT);   // B=(W2.T): k-contig
    gat(Hpre, a3_src, a3_dst, Hagg, D2);
    elu2bf(Hagg, Cat, D2, D2, 0);                             // h3 (bf16)
    gemm(Cat,      D2, nullptr, W1xbf, NH, h4x, Nn, IN_X, NH); // B=(W1x.T)
    gemm(Cat + NH, D2, nullptr, W1rbf, NH, h4r, Nn, IN_R, NH); // B=(W1r.T)
  };

  encode(nullptr, out + oh2);
  decode(out + oh2, out + oh4x, out + oh4r);
  encode(perm, out + orh2);
  decode(out + orh2, out + orh4x, out + orh4r);
  mean_sigmoid_kernel<<<1, OUT, 0, stream>>>(out + oh2, out + osum, Nn, OUT);
}
